// GPTModel_3075196584193
// MI455X (gfx1250) — compile-verified, matrix operand order checked by reading, not run on hardware
//
#include <hip/hip_runtime.h>
#include <hip/hip_bf16.h>
#include <stdint.h>
#include <stddef.h>

typedef __bf16 bf16;
typedef __attribute__((ext_vector_type(16))) __bf16 v16bf;
typedef __attribute__((ext_vector_type(8)))  float  v8f;
typedef __attribute__((ext_vector_type(4)))  float  f32x4;
typedef __attribute__((ext_vector_type(4)))  unsigned int u32x4;
typedef __attribute__((ext_vector_type(2)))  unsigned int u32x2;

#define DEV static __device__ __forceinline__

constexpr int Lc = 4, Dc = 1024, Hc = 16, DHc = 64, FFc = 4096, Vc = 50257, Tc = 1024, Bc = 2;
constexpr int BTc = Bc * Tc;

union Frag {
    v16bf v;
    uint32_t u[8];
    u32x4 q[2];
};

DEV v8f zero8() {
    v8f z;
#pragma unroll
    for (int i = 0; i < 8; ++i) z[i] = 0.0f;
    return z;
}

DEV uint32_t pack2(float a, float b) {
    union { bf16 h[2]; uint32_t u; } x;
    x.h[0] = (bf16)a;
    x.h[1] = (bf16)b;
    return x.u;
}

// ISA 16-bit fragment layout: lane (l&15) = row/col index, g = l>>4 selects the
// K-half. A lane's 16 bf16 elements are two contiguous 16B chunks at
// k = g*8 and k = 16 + g*8  ->  two b128 loads per fragment.
DEV v16bf load_frag_b16(const bf16* mat, int row, int ld, int koff, int nrows) {
    int lane = threadIdx.x & 31;
    int r = row + (lane & 15);
    int g = lane >> 4;
    bool ok = (r < nrows);
    Frag f;
    if (ok) {
        const u32x4* p = (const u32x4*)(mat + (size_t)r * ld + koff + g * 8);
        f.q[0] = p[0];   // k = g*8 .. g*8+7
        f.q[1] = p[2];   // k = 16+g*8 .. 16+g*8+7
    } else {
#pragma unroll
        for (int i = 0; i < 8; ++i) f.u[i] = 0u;
    }
    return f.v;
}

// Same fragment from a strided f32 source: four b128 loads + pack to bf16.
DEV v16bf load_frag_f32(const float* base, size_t stride, int row, int koff, float scale) {
    int lane = threadIdx.x & 31;
    int r = row + (lane & 15);
    int g = lane >> 4;
    const f32x4* p = (const f32x4*)(base + (size_t)r * stride + koff + g * 8);
    f32x4 a0 = p[0], a1 = p[1];   // k = g*8 .. +7
    f32x4 b0 = p[4], b1 = p[5];   // k = 16+g*8 .. +7
    Frag f;
    f.u[0] = pack2(a0.x * scale, a0.y * scale);
    f.u[1] = pack2(a0.z * scale, a0.w * scale);
    f.u[2] = pack2(a1.x * scale, a1.y * scale);
    f.u[3] = pack2(a1.z * scale, a1.w * scale);
    f.u[4] = pack2(b0.x * scale, b0.y * scale);
    f.u[5] = pack2(b0.z * scale, b0.w * scale);
    f.u[6] = pack2(b1.x * scale, b1.y * scale);
    f.u[7] = pack2(b1.z * scale, b1.w * scale);
    return f.v;
}

DEV v8f wmma_bf16(v16bf a, v16bf b, v8f c) {
    return __builtin_amdgcn_wmma_f32_16x16x32_bf16(false, a, false, b, (short)0, c, false, false);
}

DEV float gelu_tanh(float x) {
    const float c = 0.7978845608028654f;
    float t = c * (x + 0.044715f * x * x * x);
    return 0.5f * x * (1.0f + tanhf(t));
}

// ---------------------------------------------------------------- conversions

__global__ void cvt_bf16_kernel(const float* __restrict__ in, bf16* __restrict__ out, size_t n4) {
    // n4 = n/4, n multiple of 4
    size_t i = (size_t)blockIdx.x * blockDim.x + threadIdx.x;
    size_t stride = (size_t)gridDim.x * blockDim.x;
    for (; i < n4; i += stride) {
        f32x4 v = ((const f32x4*)in)[i];
        u32x2 o;
        o.x = pack2(v.x, v.y);
        o.y = pack2(v.z, v.w);
        ((u32x2*)out)[i] = o;
    }
}

// in: [L][K][N] f32  ->  out: [L][N][K] bf16
__global__ void transpose_cvt_kernel(const float* __restrict__ in, bf16* __restrict__ out,
                                     int K, int N) {
    __shared__ float tile[32][33];
    const float* ip = in + (size_t)blockIdx.z * K * N;
    bf16* op = out + (size_t)blockIdx.z * K * N;
    int n0 = blockIdx.x * 32, k0 = blockIdx.y * 32;
    int tx = threadIdx.x, ty = threadIdx.y;
#pragma unroll
    for (int i = 0; i < 4; ++i)
        tile[ty + 8 * i][tx] = ip[(size_t)(k0 + ty + 8 * i) * N + n0 + tx];
    __syncthreads();
#pragma unroll
    for (int i = 0; i < 4; ++i)
        op[(size_t)(n0 + ty + 8 * i) * K + k0 + tx] = (bf16)tile[tx][ty + 8 * i];
}

// ---------------------------------------------------------------- embedding

__global__ void embed_kernel(const int* __restrict__ idx, const float* __restrict__ wte,
                             const float* __restrict__ wpe, float* __restrict__ x) {
    int bt = blockIdx.x;
    int tok = idx[bt];
    int t = bt % Tc;
    const f32x4* we = (const f32x4*)(wte + (size_t)tok * Dc);
    const f32x4* pe = (const f32x4*)(wpe + (size_t)t * Dc);
    f32x4* xo = (f32x4*)(x + (size_t)bt * Dc);
    int i = threadIdx.x;  // 256 threads * 4 = 1024 = D
    xo[i] = we[i] + pe[i];
}

// ---------------------------------------------------------------- layernorm

__global__ void layernorm_kernel(const float* __restrict__ x, const float* __restrict__ gg,
                                 const float* __restrict__ bb, bf16* __restrict__ out) {
    int row = blockIdx.x, tid = threadIdx.x;
    f32x4 v = ((const f32x4*)(x + (size_t)row * Dc))[tid];
    float s = v.x + v.y + v.z + v.w;
    float s2 = v.x * v.x + v.y * v.y + v.z * v.z + v.w * v.w;
#pragma unroll
    for (int off = 16; off > 0; off >>= 1) {
        s += __shfl_xor(s, off);
        s2 += __shfl_xor(s2, off);
    }
    __shared__ float rs[8], rs2[8];
    int w = tid >> 5;
    if ((tid & 31) == 0) { rs[w] = s; rs2[w] = s2; }
    __syncthreads();
    float S = 0.0f, S2 = 0.0f;
#pragma unroll
    for (int i = 0; i < 8; ++i) { S += rs[i]; S2 += rs2[i]; }
    float mean = S * (1.0f / Dc);
    float var = S2 * (1.0f / Dc) - mean * mean;
    float rstd = rsqrtf(var + 1e-5f);
    f32x4 gv = ((const f32x4*)gg)[tid];
    f32x4 bv = ((const f32x4*)bb)[tid];
    u32x2 o;
    o.x = pack2((v.x - mean) * rstd * gv.x + bv.x, (v.y - mean) * rstd * gv.y + bv.y);
    o.y = pack2((v.z - mean) * rstd * gv.z + bv.z, (v.w - mean) * rstd * gv.w + bv.w);
    ((u32x2*)(out + (size_t)row * Dc))[tid] = o;
}

// ---------------------------------------------------------------- WMMA GEMM
// C[M,N] = A[M,K](bf16 row-major) x B[N,K](bf16 row-major = B^T storage)
// Block: 256 threads = 8 waves (4 in M x 2 in N); wave tile = 32(M) x 64(N):
// 2 A-frags + 4 B-frags -> 8 WMMAs per 32-deep K step (12 b128 loads).

template <bool BIAS, bool GELU, bool RES, bool OUTBF>
__global__ void gemm_bt_kernel(const bf16* __restrict__ A, const bf16* __restrict__ Bm,
                               const float* __restrict__ bias, const float* __restrict__ res,
                               float* __restrict__ Cf, bf16* __restrict__ Cb,
                               int M, int N, int K) {
    int w = threadIdx.x >> 5;
    int mo = blockIdx.y * 128 + (w & 3) * 32;
    int no = blockIdx.x * 128 + (w >> 2) * 64;

    v8f acc0[4], acc1[4];
#pragma unroll
    for (int t = 0; t < 4; ++t) { acc0[t] = zero8(); acc1[t] = zero8(); }

    for (int k = 0; k < K; k += 32) {
        v16bf a0 = load_frag_b16(A, mo, K, k, M);
        v16bf a1 = load_frag_b16(A, mo + 16, K, k, M);
#pragma unroll
        for (int t = 0; t < 4; ++t) {
            v16bf b = load_frag_b16(Bm, no + t * 16, K, k, N);
            acc0[t] = wmma_bf16(a0, b, acc0[t]);
            acc1[t] = wmma_bf16(a1, b, acc1[t]);
        }
    }

    int lane = threadIdx.x & 31;
    int c = lane & 15;
    int g = lane >> 4;
#pragma unroll
    for (int t = 0; t < 4; ++t) {
        int n = no + t * 16 + c;
        if (n < N) {
            float bv = BIAS ? bias[n] : 0.0f;
#pragma unroll
            for (int half = 0; half < 2; ++half) {
                v8f* acc = half ? acc1 : acc0;
#pragma unroll
                for (int r = 0; r < 8; ++r) {
                    int m = mo + half * 16 + r + 8 * g;
                    float x = acc[t][r] + bv;
                    if (GELU) x = gelu_tanh(x);
                    if (RES) x += res[(size_t)m * N + n];
                    if (OUTBF) Cb[(size_t)m * N + n] = (bf16)x;
                    else       Cf[(size_t)m * N + n] = x;
                }
            }
        }
    }
}

// ---------------------------------------------------------------- attention
// One 32-thread wave per (b, h, 16-query tile). Flash-style online softmax.

__global__ void attention_kernel(const float* __restrict__ qkv, bf16* __restrict__ yB) {
    __shared__ bf16 Pl[16 * 32];  // P tile (q x 32 keys), row-major for A-frag reload
    __shared__ bf16 Vt[64 * 32];  // V^T tile: [dh][key]

    const int lane = threadIdx.x & 31;
    const int c = lane & 15;
    const int g = lane >> 4;
    const int qo = blockIdx.x * 16;
    const int b = blockIdx.y >> 4;
    const int h = blockIdx.y & 15;
    const size_t row3D = (size_t)3 * Dc;
    const float* qbase = qkv + (size_t)b * Tc * row3D + (size_t)h * DHc;
    const float* kbase = qbase + Dc;
    const float* vbase = qbase + 2 * Dc;
    const float scale = 0.125f;  // 1/sqrt(64)

    v16bf qa0 = load_frag_f32(qbase, row3D, qo, 0, scale);
    v16bf qa1 = load_frag_f32(qbase, row3D, qo, 32, scale);

    v8f o[4];
#pragma unroll
    for (int t = 0; t < 4; ++t) o[t] = zero8();
    float m_i[8], l_i[8];
#pragma unroll
    for (int r = 0; r < 8; ++r) { m_i[r] = -3.0e38f; l_i[r] = 0.0f; }

    const int kend = qo + 16;
    for (int kj = 0; kj < kend; kj += 32) {
        v8f s0 = zero8(), s1 = zero8();
        {
            v16bf kb = load_frag_f32(kbase, row3D, kj, 0, 1.0f);
            s0 = wmma_bf16(qa0, kb, s0);
            kb = load_frag_f32(kbase, row3D, kj, 32, 1.0f);
            s0 = wmma_bf16(qa1, kb, s0);
            kb = load_frag_f32(kbase, row3D, kj + 16, 0, 1.0f);
            s1 = wmma_bf16(qa0, kb, s1);
            kb = load_frag_f32(kbase, row3D, kj + 16, 32, 1.0f);
            s1 = wmma_bf16(qa1, kb, s1);
        }

#pragma unroll
        for (int r = 0; r < 8; ++r) {
            int m = qo + r + 8 * g;
            if (kj + c > m) s0[r] = -1.0e30f;
            if (kj + 16 + c > m) s1[r] = -1.0e30f;

            float rmax = fmaxf(s0[r], s1[r]);
#pragma unroll
            for (int off = 1; off < 16; off <<= 1) rmax = fmaxf(rmax, __shfl_xor(rmax, off));
            float mnew = fmaxf(m_i[r], rmax);
            float alpha = __expf(m_i[r] - mnew);
            float p0 = __expf(s0[r] - mnew);
            float p1 = __expf(s1[r] - mnew);
            float rsum = p0 + p1;
#pragma unroll
            for (int off = 1; off < 16; off <<= 1) rsum += __shfl_xor(rsum, off);
            l_i[r] = l_i[r] * alpha + rsum;
            m_i[r] = mnew;
#pragma unroll
            for (int t = 0; t < 4; ++t) o[t][r] *= alpha;
            Pl[(r + 8 * g) * 32 + c] = (bf16)p0;
            Pl[(r + 8 * g) * 32 + 16 + c] = (bf16)p1;
        }

        // Stage V^T: lane loads one key row (64 dh) with b128 loads.
        {
            const f32x4* vp = (const f32x4*)(vbase + (size_t)(kj + lane) * row3D);
#pragma unroll
            for (int i = 0; i < 16; ++i) {
                f32x4 v4 = vp[i];
                Vt[(4 * i + 0) * 32 + lane] = (bf16)v4.x;
                Vt[(4 * i + 1) * 32 + lane] = (bf16)v4.y;
                Vt[(4 * i + 2) * 32 + lane] = (bf16)v4.z;
                Vt[(4 * i + 3) * 32 + lane] = (bf16)v4.w;
            }
        }
        __syncthreads();

        v16bf pa = load_frag_b16(Pl, 0, 32, 0, 16);
#pragma unroll
        for (int t = 0; t < 4; ++t) {
            v16bf vb = load_frag_b16(Vt, t * 16, 32, 0, 64);
            o[t] = wmma_bf16(pa, vb, o[t]);
        }
        __syncthreads();
    }

    float inv[8];
#pragma unroll
    for (int r = 0; r < 8; ++r) inv[r] = 1.0f / l_i[r];
#pragma unroll
    for (int t = 0; t < 4; ++t) {
        int dh = t * 16 + c;
#pragma unroll
        for (int r = 0; r < 8; ++r) {
            int m = qo + r + 8 * g;
            yB[(size_t)(b * Tc + m) * Dc + h * DHc + dh] = (bf16)(o[t][r] * inv[r]);
        }
    }
}

// ---------------------------------------------------------------- loss

__global__ void nll_kernel(const float* __restrict__ logits, const int* __restrict__ targets,
                           float* __restrict__ nll) {
    int row = blockIdx.x, tid = threadIdx.x;
    const float* x = logits + (size_t)row * Vc;
    float m = -3.0e38f, l = 0.0f;
    for (int j = tid; j < Vc; j += 256) {
        float v = x[j];
        float mn = fmaxf(m, v);
        l = l * __expf(m - mn) + __expf(v - mn);
        m = mn;
    }
    __shared__ float sm[256], sl[256];
    sm[tid] = m;
    sl[tid] = l;
    __syncthreads();
    for (int s = 128; s > 0; s >>= 1) {
        if (tid < s) {
            float m2 = sm[tid + s], l2 = sl[tid + s];
            float mn = fmaxf(sm[tid], m2);
            sl[tid] = sl[tid] * __expf(sm[tid] - mn) + l2 * __expf(m2 - mn);
            sm[tid] = mn;
        }
        __syncthreads();
    }
    if (tid == 0) {
        int t = targets[row];
        nll[row] = sm[0] + __logf(sl[0]) - x[t];
    }
}

__global__ void loss_reduce_kernel(const float* __restrict__ nll, float* __restrict__ out) {
    float s = 0.0f;
    for (int j = threadIdx.x; j < BTc; j += 256) s += nll[j];
    __shared__ float ss[256];
    ss[threadIdx.x] = s;
    __syncthreads();
    for (int k = 128; k > 0; k >>= 1) {
        if (threadIdx.x < k) ss[threadIdx.x] += ss[threadIdx.x + k];
        __syncthreads();
    }
    if (threadIdx.x == 0) out[0] = ss[0] * (1.0f / BTc);
}

// ---------------------------------------------------------------- launch

extern "C" void kernel_launch(void* const* d_in, const int* in_sizes, int n_in,
                              void* d_out, int out_size, void* d_ws, size_t ws_size,
                              hipStream_t stream) {
    (void)in_sizes; (void)n_in; (void)out_size; (void)ws_size;

    const int* idx = (const int*)d_in[0];
    const int* targets = (const int*)d_in[1];
    const float* wte = (const float*)d_in[2];
    const float* wpe = (const float*)d_in[3];
    const float* ln1_g = (const float*)d_in[4];
    const float* ln1_b = (const float*)d_in[5];
    const float* ln2_g = (const float*)d_in[6];
    const float* ln2_b = (const float*)d_in[7];
    const float* Wqkv = (const float*)d_in[8];
    const float* Wo = (const float*)d_in[9];
    const float* W1 = (const float*)d_in[10];
    const float* b1 = (const float*)d_in[11];
    const float* W2 = (const float*)d_in[12];
    const float* b2 = (const float*)d_in[13];
    const float* lnf_g = (const float*)d_in[14];
    const float* lnf_b = (const float*)d_in[15];

    char* ws = (char*)d_ws;
    size_t off = 0;
    auto alloc = [&](size_t bytes) {
        size_t r = off;
        off = (off + bytes + 255) & ~(size_t)255;
        return r;
    };
    bf16* wteB  = (bf16*)(ws + alloc((size_t)Vc * Dc * 2));
    bf16* wqkvB = (bf16*)(ws + alloc((size_t)Lc * 3 * Dc * Dc * 2));  // [L][3D][D]
    bf16* woB   = (bf16*)(ws + alloc((size_t)Lc * Dc * Dc * 2));      // [L][D][D]
    bf16* w1B   = (bf16*)(ws + alloc((size_t)Lc * FFc * Dc * 2));     // [L][FF][D]
    bf16* w2B   = (bf16*)(ws + alloc((size_t)Lc * Dc * FFc * 2));     // [L][D][FF]
    float* xf   = (float*)(ws + alloc((size_t)BTc * Dc * 4));
    bf16* hB    = (bf16*)(ws + alloc((size_t)BTc * Dc * 2));
    float* qkvf = (float*)(ws + alloc((size_t)BTc * 3 * Dc * 4));
    bf16* yB    = (bf16*)(ws + alloc((size_t)BTc * Dc * 2));
    bf16* ffB   = (bf16*)(ws + alloc((size_t)BTc * FFc * 2));
    float* nllf = (float*)(ws + alloc((size_t)BTc * 4));

    float* logits = (float*)d_out;
    float* lossp = logits + (size_t)BTc * Vc;

    // 1) Weights -> bf16 [N,K] layouts (one pass, ~407MB read @ 23.3 TB/s).
    cvt_bf16_kernel<<<4096, 256, 0, stream>>>(wte, wteB, (size_t)Vc * Dc / 4);
    transpose_cvt_kernel<<<dim3(3 * Dc / 32, Dc / 32, Lc), dim3(32, 8), 0, stream>>>(Wqkv, wqkvB, Dc, 3 * Dc);
    transpose_cvt_kernel<<<dim3(Dc / 32, Dc / 32, Lc), dim3(32, 8), 0, stream>>>(Wo, woB, Dc, Dc);
    transpose_cvt_kernel<<<dim3(FFc / 32, Dc / 32, Lc), dim3(32, 8), 0, stream>>>(W1, w1B, Dc, FFc);
    transpose_cvt_kernel<<<dim3(Dc / 32, FFc / 32, Lc), dim3(32, 8), 0, stream>>>(W2, w2B, FFc, Dc);

    // 2) Embedding
    embed_kernel<<<BTc, 256, 0, stream>>>(idx, wte, wpe, xf);

    // 3) Transformer layers
    for (int l = 0; l < Lc; ++l) {
        layernorm_kernel<<<BTc, 256, 0, stream>>>(xf, ln1_g + (size_t)l * Dc, ln1_b + (size_t)l * Dc, hB);

        gemm_bt_kernel<false, false, false, false>
            <<<dim3(3 * Dc / 128, BTc / 128), 256, 0, stream>>>(
                hB, wqkvB + (size_t)l * 3 * Dc * Dc, nullptr, nullptr, qkvf, nullptr,
                BTc, 3 * Dc, Dc);

        attention_kernel<<<dim3(Tc / 16, Bc * Hc), 32, 0, stream>>>(qkvf, yB);

        gemm_bt_kernel<false, false, true, false>
            <<<dim3(Dc / 128, BTc / 128), 256, 0, stream>>>(
                yB, woB + (size_t)l * Dc * Dc, nullptr, xf, xf, nullptr, BTc, Dc, Dc);

        layernorm_kernel<<<BTc, 256, 0, stream>>>(xf, ln2_g + (size_t)l * Dc, ln2_b + (size_t)l * Dc, hB);

        gemm_bt_kernel<true, true, false, true>
            <<<dim3(FFc / 128, BTc / 128), 256, 0, stream>>>(
                hB, w1B + (size_t)l * FFc * Dc, b1 + (size_t)l * FFc, nullptr, nullptr, ffB,
                BTc, FFc, Dc);

        gemm_bt_kernel<true, false, true, false>
            <<<dim3(Dc / 128, BTc / 128), 256, 0, stream>>>(
                ffB, w2B + (size_t)l * Dc * FFc, b2 + (size_t)l * Dc, xf, xf, nullptr,
                BTc, Dc, FFc);
    }

    // 4) Final LN + tied lm_head (wte already stored [V, D] = [N, K])
    layernorm_kernel<<<BTc, 256, 0, stream>>>(xf, lnf_g, lnf_b, hB);
    gemm_bt_kernel<false, false, false, false>
        <<<dim3((Vc + 127) / 128, BTc / 128), 256, 0, stream>>>(
            hB, wteB, nullptr, nullptr, logits, nullptr, BTc, Vc, Dc);

    // 5) NLL + mean loss
    nll_kernel<<<BTc, 256, 0, stream>>>(logits, targets, nllf);
    loss_reduce_kernel<<<1, 256, 0, stream>>>(nllf, lossp);
}
